// GINBaseline_84232898609756
// MI455X (gfx1250) — matrix operand order repaired and checked by reading, not compiled, not run on hardware
//
#include <hip/hip_runtime.h>
#include <hip/hip_bf16.h>

#define NN_NODES 100000
#define NN_EDGES 200000
#define NN_GRAPHS 64
#define FF_IN 5
#define HID 768
#define EMB 512
#define NN_LAYERS 16

typedef unsigned short u16;
typedef unsigned int u32;
typedef __attribute__((ext_vector_type(16))) __bf16 v16bf;
typedef __attribute__((ext_vector_type(8)))  float  v8f;

union Frag16 { v16bf v; uint4 q[2]; };

__device__ __forceinline__ u16 f2bf(float f) {
    u32 x = __float_as_uint(f);
    u32 r = x + 0x7fffu + ((x >> 16) & 1u);   // round-to-nearest-even
    return (u16)(r >> 16);
}

// ---------------------------------------------------------------------------
// Weight conversion: W1/W2 [16][K=768][N=768] fp32 -> bf16 transposed [32][N][K]
// ---------------------------------------------------------------------------
__global__ void convert_w12(const float* __restrict__ W1, const float* __restrict__ W2,
                            u16* __restrict__ wt) {
    int i = blockIdx.x * blockDim.x + threadIdx.x;
    const int tot = 2 * NN_LAYERS * HID * HID;
    if (i >= tot) return;
    int l2  = i / (HID * HID);
    int rem = i - l2 * (HID * HID);
    int k = rem / HID;
    int n = rem - k * HID;
    const float* src = (l2 < NN_LAYERS) ? (W1 + (size_t)l2 * HID * HID)
                                        : (W2 + (size_t)(l2 - NN_LAYERS) * HID * HID);
    wt[(size_t)l2 * HID * HID + (size_t)n * HID + k] = f2bf(src[(size_t)k * HID + n]);
}

// Generic [K][N] fp32 -> [N][K] bf16
__global__ void convert_tr(const float* __restrict__ W, u16* __restrict__ Wt,
                           int Kd, int Nd) {
    int i = blockIdx.x * blockDim.x + threadIdx.x;
    if (i >= Kd * Nd) return;
    int k = i / Nd, n = i - k * Nd;
    Wt[(size_t)n * Kd + k] = f2bf(W[i]);
}

// ---------------------------------------------------------------------------
// Input projection: h = x @ Wp + bp   ([N,5] @ [5,768])
// ---------------------------------------------------------------------------
__global__ void input_proj(const float* __restrict__ x, const float* __restrict__ Wp,
                           const float* __restrict__ bp, float* __restrict__ h) {
    int tid = blockIdx.x * blockDim.x + threadIdx.x;
    if (tid >= NN_NODES * HID) return;
    int n = tid / HID, j = tid - n * HID;
    float acc = bp[j];
#pragma unroll
    for (int f = 0; f < FF_IN; ++f) acc += x[n * FF_IN + f] * Wp[f * HID + j];
    h[tid] = acc;
}

// ---------------------------------------------------------------------------
// agg = h  (vectorized copy), then edge scatter-add: agg[dst] += h[src]
// ---------------------------------------------------------------------------
__global__ void copy_f4(const float4* __restrict__ s, float4* __restrict__ d, int n4) {
    int i = blockIdx.x * blockDim.x + threadIdx.x;
    if (i < n4) d[i] = s[i];
}

__global__ void scatter_edges(const float* __restrict__ h, const int* __restrict__ src,
                              const int* __restrict__ dst, float* __restrict__ agg) {
    const int perEdge = HID / 4;
    int tid = blockIdx.x * blockDim.x + threadIdx.x;
    if (tid >= NN_EDGES * perEdge) return;
    int e = tid / perEdge;
    int c = (tid - e * perEdge) * 4;
    int s = src[e], d = dst[e];
    float4 v = *reinterpret_cast<const float4*>(&h[(size_t)s * HID + c]);
    float* o = &agg[(size_t)d * HID + c];
    atomicAdd(o + 0, v.x); atomicAdd(o + 1, v.y);
    atomicAdd(o + 2, v.z); atomicAdd(o + 3, v.w);
}

// ---------------------------------------------------------------------------
// WMMA GEMM: C[M,N] = act(A[M,K](fp32) @ Bt[N,K](bf16)^T + bias)
// 256 threads (8 waves), block tile 128x128, K-step 32, LDS double-buffered.
// ---------------------------------------------------------------------------
#define MT 128
#define NT 128
#define KT 32
#define LDT 48   // LDS row stride in u16 (96B: 16B aligned, bank-friendly)

__global__ __launch_bounds__(256)
void gemm_bf16_wmma(const float* __restrict__ A, const u16* __restrict__ Bt,
                    const float* __restrict__ bias, float* __restrict__ C,
                    int M, int N, int K, int doRelu) {
    __shared__ __align__(16) u16 sA[2][MT * LDT];
    __shared__ __align__(16) u16 sB[2][NT * LDT];

    const int tid  = threadIdx.x;
    const int lane = tid & 31;
    const int wave = tid >> 5;        // 0..7, owns rows [wave*16, wave*16+16)
    const int mb   = blockIdx.x;
    const int nb   = blockIdx.y;
    const int half = lane >> 4;       // 0 / 1
    const int l16  = lane & 15;

    v8f acc[8];
#pragma unroll
    for (int t = 0; t < 8; ++t)
#pragma unroll
        for (int i = 0; i < 8; ++i) acc[t][i] = 0.0f;

    // ---- per-thread loader roles (fixed across K-steps) ----
    // A: 4 segments of 4 floats; B: 2 segments of 8 bf16
    int  rA[4], cA[4];  bool okA[4];  const float* pA[4];
    int  rB[2], cB[2];  const u16*   pB[2];
#pragma unroll
    for (int i = 0; i < 4; ++i) {
        int seg = tid * 4 + i;
        rA[i] = seg >> 3;
        cA[i] = (seg & 7) * 4;
        int m = mb * MT + rA[i];
        okA[i] = (m < M);
        pA[i] = A + (size_t)(okA[i] ? m : 0) * K + cA[i];
    }
#pragma unroll
    for (int i = 0; i < 2; ++i) {
        int seg = tid * 2 + i;
        rB[i] = seg >> 2;
        cB[i] = (seg & 3) * 8;
        pB[i] = Bt + (size_t)(nb * NT + rB[i]) * K + cB[i];
    }

    // A-fragment LDS offsets (ISA 16-bit A 16x32 layout):
    //  lane<16: elems 0..7 = K0..7, elems 8..15 = K16..23 ; lane>=16: +8
    const int aRow   = wave * 16 + l16;
    const int aOffLo = aRow * LDT + half * 8;
    const int aOffHi = aRow * LDT + 16 + half * 8;

    const int steps = K / KT;

    float4 va[4];
    uint4  vb[2];

    // prologue: fetch k-step 0 and stage into buffer 0
#pragma unroll
    for (int i = 0; i < 4; ++i)
        va[i] = okA[i] ? *reinterpret_cast<const float4*>(pA[i])
                       : make_float4(0.f, 0.f, 0.f, 0.f);
#pragma unroll
    for (int i = 0; i < 2; ++i)
        vb[i] = *reinterpret_cast<const uint4*>(pB[i]);
#pragma unroll
    for (int i = 0; i < 4; ++i) {
        u16* p = &sA[0][rA[i] * LDT + cA[i]];
        p[0] = f2bf(va[i].x); p[1] = f2bf(va[i].y);
        p[2] = f2bf(va[i].z); p[3] = f2bf(va[i].w);
    }
#pragma unroll
    for (int i = 0; i < 2; ++i)
        *reinterpret_cast<uint4*>(&sB[0][rB[i] * LDT + cB[i]]) = vb[i];

    for (int it = 0; it < steps; ++it) {
        const int cur = it & 1;
        __syncthreads();

        // issue next tile's global loads; they overlap the WMMA chain below
        if (it + 1 < steps) {
            const int k1 = (it + 1) * KT;
#pragma unroll
            for (int i = 0; i < 4; ++i)
                va[i] = okA[i] ? *reinterpret_cast<const float4*>(pA[i] + k1)
                               : make_float4(0.f, 0.f, 0.f, 0.f);
#pragma unroll
            for (int i = 0; i < 2; ++i)
                vb[i] = *reinterpret_cast<const uint4*>(pB[i] + k1);
        }
        if (it + 2 < steps) {
            const int k2 = (it + 2) * KT;
            __builtin_prefetch(pA[0] + k2, 0, 1);   // global_prefetch_b8
            __builtin_prefetch(pB[0] + k2, 0, 1);
        }

        // ---- fragments: issue all LDS loads first, then WMMA back-to-back ----
        Frag16 fa;
        fa.q[0] = *reinterpret_cast<const uint4*>(&sA[cur][aOffLo]);
        fa.q[1] = *reinterpret_cast<const uint4*>(&sA[cur][aOffHi]);

        Frag16 fb[8];
#pragma unroll
        for (int t = 0; t < 8; ++t) {
            const int bOff = (t * 16 + l16) * LDT + half * 8;
            fb[t].q[0] = *reinterpret_cast<const uint4*>(&sB[cur][bOff]);
            fb[t].q[1] = *reinterpret_cast<const uint4*>(&sB[cur][bOff + 16]);
        }
#pragma unroll
        for (int t = 0; t < 8; ++t)
            acc[t] = __builtin_amdgcn_wmma_f32_16x16x32_bf16(
                         false, fa.v, false, fb[t].v, (short)0, acc[t], false, false);

        // stage next tile into the other buffer (safe: everyone passed the
        // barrier above, so nobody is still reading buffer cur^1)
        if (it + 1 < steps) {
            const int nxt = cur ^ 1;
#pragma unroll
            for (int i = 0; i < 4; ++i) {
                u16* p = &sA[nxt][rA[i] * LDT + cA[i]];
                p[0] = f2bf(va[i].x); p[1] = f2bf(va[i].y);
                p[2] = f2bf(va[i].z); p[3] = f2bf(va[i].w);
            }
#pragma unroll
            for (int i = 0; i < 2; ++i)
                *reinterpret_cast<uint4*>(&sB[nxt][rB[i] * LDT + cB[i]]) = vb[i];
        }
    }

    // ---- epilogue: bias + ReLU + store (C/D layout: M = r + 8*half, N = l16) ----
#pragma unroll
    for (int t = 0; t < 8; ++t) {
        int n  = nb * NT + t * 16 + l16;
        float bv = bias[n];
#pragma unroll
        for (int r = 0; r < 8; ++r) {
            int m = mb * MT + wave * 16 + half * 8 + r;
            if (m < M) {
                float v = acc[t][r] + bv;
                if (doRelu) v = fmaxf(v, 0.0f);
                C[(size_t)m * N + n] = v;
            }
        }
    }
}

// ---------------------------------------------------------------------------
// Pooling
// ---------------------------------------------------------------------------
__global__ void zero_f(float* p, int n) {
    int i = blockIdx.x * blockDim.x + threadIdx.x;
    if (i < n) p[i] = 0.0f;
}

__global__ void pool_sum(const float* __restrict__ h, const int* __restrict__ batch,
                         float* __restrict__ sums) {
    const int perNode = HID / 4;
    int tid = blockIdx.x * blockDim.x + threadIdx.x;
    if (tid >= NN_NODES * perNode) return;
    int n = tid / perNode;
    int c = (tid - n * perNode) * 4;
    int g = batch[n];
    float4 v = *reinterpret_cast<const float4*>(&h[(size_t)n * HID + c]);
    float* o = &sums[(size_t)g * HID + c];
    atomicAdd(o + 0, v.x); atomicAdd(o + 1, v.y);
    atomicAdd(o + 2, v.z); atomicAdd(o + 3, v.w);
}

__global__ void count_hist(const int* __restrict__ batch, float* __restrict__ counts) {
    __shared__ int hist[NN_GRAPHS];
    if (threadIdx.x < NN_GRAPHS) hist[threadIdx.x] = 0;
    __syncthreads();
    int n = blockIdx.x * blockDim.x + threadIdx.x;
    if (n < NN_NODES) atomicAdd(&hist[batch[n]], 1);
    __syncthreads();
    if (threadIdx.x < NN_GRAPHS && hist[threadIdx.x] != 0)
        atomicAdd(&counts[threadIdx.x], (float)hist[threadIdx.x]);
}

__global__ void pool_div(const float* __restrict__ sums, const float* __restrict__ counts,
                         float* __restrict__ pooled) {
    int i = blockIdx.x * blockDim.x + threadIdx.x;
    if (i >= NN_GRAPHS * HID) return;
    int g = i / HID;
    pooled[i] = sums[i] / fmaxf(counts[g], 1.0f);
}

// ---------------------------------------------------------------------------
// Host-side orchestration
// ---------------------------------------------------------------------------
extern "C" void kernel_launch(void* const* d_in, const int* in_sizes, int n_in,
                              void* d_out, int out_size, void* d_ws, size_t ws_size,
                              hipStream_t stream) {
    const float* x     = (const float*)d_in[0];
    const int*   ei    = (const int*)d_in[1];
    const int*   batch = (const int*)d_in[2];
    const float* Wp    = (const float*)d_in[3];
    const float* bp    = (const float*)d_in[4];
    const float* W1    = (const float*)d_in[5];
    const float* b1    = (const float*)d_in[6];
    const float* W2    = (const float*)d_in[7];
    const float* b2    = (const float*)d_in[8];
    const float* Wf    = (const float*)d_in[9];
    const float* bfb   = (const float*)d_in[10];
    float* out = (float*)d_out;

    char* ws = (char*)d_ws;
    size_t off = 0;
    auto take = [&](size_t bytes) -> char* {
        char* p = ws + off;
        off += (bytes + 255) & ~(size_t)255;
        return p;
    };
    float* h      = (float*)take((size_t)NN_NODES * HID * 4);
    float* agg    = (float*)take((size_t)NN_NODES * HID * 4);
    float* z      = (float*)take((size_t)NN_NODES * HID * 4);
    u16*   wt12   = (u16*)  take((size_t)2 * NN_LAYERS * HID * HID * 2);
    u16*   wtf    = (u16*)  take((size_t)EMB * HID * 2);
    float* sums   = (float*)take((size_t)NN_GRAPHS * HID * 4);
    float* pooled = (float*)take((size_t)NN_GRAPHS * HID * 4);
    float* counts = (float*)take(256);

    const int* srcE = ei;
    const int* dstE = ei + NN_EDGES;
    const int B = 256;

    // weights -> bf16 transposed (fits in 192MB L2, reused by all 32 GEMM passes)
    { int tot = 2 * NN_LAYERS * HID * HID;
      convert_w12<<<(tot + B - 1) / B, B, 0, stream>>>(W1, W2, wt12); }
    { int tot = HID * EMB;
      convert_tr<<<(tot + B - 1) / B, B, 0, stream>>>(Wf, wtf, HID, EMB); }

    // input projection
    { int tot = NN_NODES * HID;
      input_proj<<<(tot + B - 1) / B, B, 0, stream>>>(x, Wp, bp, h); }

    dim3 gemmGrid((NN_NODES + MT - 1) / MT, HID / NT);
    const int n4 = NN_NODES * HID / 4;
    const int st = NN_EDGES * (HID / 4);
    for (int l = 0; l < NN_LAYERS; ++l) {
        copy_f4<<<(n4 + B - 1) / B, B, 0, stream>>>((const float4*)h, (float4*)agg, n4);
        scatter_edges<<<(st + B - 1) / B, B, 0, stream>>>(h, srcE, dstE, agg);
        gemm_bf16_wmma<<<gemmGrid, B, 0, stream>>>(
            agg, wt12 + (size_t)l * HID * HID, b1 + l * HID, z,
            NN_NODES, HID, HID, 1);
        gemm_bf16_wmma<<<gemmGrid, B, 0, stream>>>(
            z, wt12 + (size_t)(NN_LAYERS + l) * HID * HID, b2 + l * HID, h,
            NN_NODES, HID, HID, 1);
    }

    // mean pool
    zero_f<<<(NN_GRAPHS * HID + B - 1) / B, B, 0, stream>>>(sums, NN_GRAPHS * HID);
    zero_f<<<1, B, 0, stream>>>(counts, NN_GRAPHS);
    { int tot = NN_NODES * (HID / 4);
      pool_sum<<<(tot + B - 1) / B, B, 0, stream>>>(h, batch, sums); }
    count_hist<<<(NN_NODES + B - 1) / B, B, 0, stream>>>(batch, counts);
    pool_div<<<(NN_GRAPHS * HID + B - 1) / B, B, 0, stream>>>(sums, counts, pooled);

    // final projection: [64,768] @ [768,512] + bf  (no ReLU)
    dim3 fGrid((NN_GRAPHS + MT - 1) / MT, EMB / NT);
    gemm_bf16_wmma<<<fGrid, B, 0, stream>>>(pooled, wtf, bfb, out,
                                            NN_GRAPHS, EMB, HID, 0);
}
